// SpatialRangeAttention_50251117363409
// MI455X (gfx1250) — matrix operand order repaired and checked by reading, not compiled, not run on hardware
//
#include <hip/hip_runtime.h>
#include <hip/hip_bf16.h>

typedef __attribute__((ext_vector_type(16))) _Float16 v16h;
typedef __attribute__((ext_vector_type(8)))  _Float16 v8h;
typedef __attribute__((ext_vector_type(8)))  float    v8f;

#define BQ    4
#define HH    64
#define WWD   64
#define HWPIX 4096      // 64*64
#define CDIM  128
#define D2K   49
#define KCAT  192       // 49+128 padded to mult of 32
#define MFX   64        // 49 padded to mult of 16

// ---------------------------------------------------------------------------
// Weight pad+convert: W[M,K] f32 -> Wp[Mp,Kp] f16, zero-padded
// ---------------------------------------------------------------------------
__global__ void pad_weight_f16(const float* __restrict__ W, int M, int K,
                               _Float16* __restrict__ Wp, int Mp, int Kp) {
  int i = blockIdx.x * blockDim.x + threadIdx.x;
  if (i >= Mp * Kp) return;
  int m = i / Kp, k = i % Kp;
  float v = (m < M && k < K) ? W[m * K + k] : 0.0f;
  Wp[i] = (_Float16)v;
}

__global__ void pad_bias_f32(const float* __restrict__ b, int len,
                             float* __restrict__ out, int Mp) {
  int i = blockIdx.x * blockDim.x + threadIdx.x;
  if (i < Mp) out[i] = (i < len) ? b[i] : 0.0f;
}

__global__ void cvt_f32_f16(const float* __restrict__ in, _Float16* __restrict__ out, int n) {
  int i = blockIdx.x * blockDim.x + threadIdx.x;
  if (i < n) out[i] = (_Float16)in[i];
}

// ---------------------------------------------------------------------------
// WMMA GEMM:  Y[b, m, p] = sum_k Wh[m,k] * X[b, k, p] + biasPad[m]
//   Wh: [Mp, KP] f16 row-major.  X: f16 [b, KP, HW].  Y: f32 [b, Mp, HW].
// Block = 4 waves sharing one 16-column panel of X staged transposed in LDS.
// Each wave computes MT M-subtiles (shared B fragment), K fully unrolled.
// Fragment layouts per CDNA5 ISA 7.12.2 (wave32):
//   A: lane L -> row M=L%16; half e -> K = e + 8g + (e>=8 ? 8 : 0), g=L/16
//   B: lane L -> col N=L%16; half e -> K = e + 8g + (e>=8 ? 8 : 0), same as A
//   D: VGPR v -> M = v + 8g, N = L%16
// ---------------------------------------------------------------------------
template <int KP, int MT>
__global__ void gemm_wmma(const _Float16* __restrict__ Wh,
                          const _Float16* __restrict__ X,
                          const float* __restrict__ biasPad,
                          float* __restrict__ Y,
                          long xbs, long ybs) {
  __shared__ _Float16 lds[16 * KP];          // transposed panel: [col j][k]
  const int lane = threadIdx.x;              // 0..31
  const int ty   = threadIdx.y;              // 0..3 (wave id)
  const int tid  = ty * 32 + lane;
  const int g = lane >> 4;
  const int r = lane & 15;
  const int n0 = blockIdx.x * 16;
  const int b  = blockIdx.z;
  const _Float16* Xb = X + (long)b * xbs;

  // cooperative staging: LDS[j*KP + k] = X[k, n0+j]
  for (int idx = tid; idx < KP * 16; idx += 128) {
    int k = idx >> 4, j = idx & 15;
    lds[j * KP + k] = Xb[(long)k * HWPIX + n0 + j];
  }
  __syncthreads();

  v8f acc[MT];
#pragma unroll
  for (int mt = 0; mt < MT; ++mt) acc[mt] = (v8f){};

#pragma unroll
  for (int k0 = 0; k0 < KP; k0 += 32) {
    // B fragment: K chunks [8g..8g+7], [16+8g..16+8g+7] from transposed LDS -> 2x ds_load_b128
    const v8h* pb = (const v8h*)(lds + r * KP + k0 + 8 * g);
    v8h bl = pb[0], bh = pb[2];
    v16h bfrag = __builtin_shufflevector(bl, bh, 0, 1, 2, 3, 4, 5, 6, 7,
                                         8, 9, 10, 11, 12, 13, 14, 15);
#pragma unroll
    for (int mt = 0; mt < MT; ++mt) {
      int m = (ty * MT + mt) * 16 + r;
      // A fragment: two contiguous 8-half chunks -> 2x global_load_b128
      const v8h* pa = (const v8h*)(Wh + (long)m * KP + k0 + 8 * g);
      v8h al = pa[0], ah = pa[2];            // K chunks [8g..8g+7], [16+8g..16+8g+7]
      v16h afrag = __builtin_shufflevector(al, ah, 0, 1, 2, 3, 4, 5, 6, 7,
                                           8, 9, 10, 11, 12, 13, 14, 15);
      acc[mt] = __builtin_amdgcn_wmma_f32_16x16x32_f16(false, afrag, false, bfrag,
                                                       (short)0, acc[mt], false, false);
    }
  }

  float* Yb = Y + (long)b * ybs;
#pragma unroll
  for (int mt = 0; mt < MT; ++mt) {
    int m0 = (ty * MT + mt) * 16;
#pragma unroll
    for (int v = 0; v < 8; ++v) {
      int m = m0 + 8 * g + v;
      Yb[(long)m * HWPIX + n0 + r] = acc[mt][v] + biasPad[m];
    }
  }
}

// ---------------------------------------------------------------------------
// Per-pixel channel LayerNorm (+ optional SiLU), f32 in -> f16 out (padded)
// ---------------------------------------------------------------------------
__global__ void ln2d_kernel(const float* __restrict__ X,   // [B, Cstride, HW]
                            const float* __restrict__ lnw,
                            const float* __restrict__ lnb,
                            _Float16* __restrict__ out,    // [B, CoutPad, HW]
                            int Cin, int Cstride, int CoutPad, int do_silu) {
  int idx = blockIdx.x * blockDim.x + threadIdx.x;
  if (idx >= BQ * HWPIX) return;
  int b = idx / HWPIX, p = idx % HWPIX;
  const float* Xb = X + (long)b * Cstride * HWPIX + p;
  float s = 0.f, s2 = 0.f;
  for (int c = 0; c < Cin; ++c) {
    float v = Xb[(long)c * HWPIX];
    s += v; s2 += v * v;
  }
  float inv = 1.0f / (float)Cin;
  float u = s * inv;
  float var = s2 * inv - u * u;
  float rstd = rsqrtf(var + 1e-6f);
  _Float16* Ob = out + (long)b * CoutPad * HWPIX + p;
  for (int c = 0; c < Cin; ++c) {
    float xn = (Xb[(long)c * HWPIX] - u) * rstd;
    float y = lnw[c] * xn + lnb[c];
    if (do_silu) y = y / (1.0f + __expf(-y));
    Ob[(long)c * HWPIX] = (_Float16)y;
  }
  for (int c = Cin; c < CoutPad; ++c) Ob[(long)c * HWPIX] = (_Float16)0.0f;
}

// ---------------------------------------------------------------------------
// Fill concat rows 49..176 with f16 semantic feats, rows 177..191 with zero
// ---------------------------------------------------------------------------
__global__ void build_concat(const float* __restrict__ sem,
                             _Float16* __restrict__ cat) {
  int idx = blockIdx.x * blockDim.x + threadIdx.x;
  if (idx >= BQ * HWPIX) return;
  int b = idx / HWPIX, p = idx % HWPIX;
  const float* Sb = sem + (long)b * CDIM * HWPIX + p;
  _Float16* Cb = cat + (long)b * KCAT * HWPIX + p;
  for (int c = 0; c < CDIM; ++c)
    Cb[(long)(D2K + c) * HWPIX] = (_Float16)Sb[(long)c * HWPIX];
  for (int c = D2K + CDIM; c < KCAT; ++c)
    Cb[(long)c * HWPIX] = (_Float16)0.0f;
}

__device__ __forceinline__ int reflect_idx(int x, int n) {
  x = (x < 0) ? -x : x;
  x = (x >= n) ? (2 * n - 2 - x) : x;
  return x;
}

// ---------------------------------------------------------------------------
// sim = <patch, center>/sqrt(DIM); softmax over 49; * gaussian spatial kernel
// writes combined f32 and combined f16 into concat rows 0..48
// ---------------------------------------------------------------------------
__global__ void sim_softmax(const _Float16* __restrict__ px, // [B,128,HW]
                            const float* __restrict__ sigma_p,
                            float* __restrict__ comb,        // [B,49,HW]
                            _Float16* __restrict__ cat) {    // [B,192,HW]
  int idx = blockIdx.x * blockDim.x + threadIdx.x;
  if (idx >= BQ * HWPIX) return;
  int b = idx / HWPIX, p = idx % HWPIX;
  int h = p / WWD, w = p % WWD;

  int pn[D2K];
#pragma unroll
  for (int n = 0; n < D2K; ++n) {
    int ki = n / 7, kj = n % 7;
    int hh = reflect_idx(h + ki - 3, HH);
    int ww = reflect_idx(w + kj - 3, WWD);
    pn[n] = hh * WWD + ww;
  }

  float sim[D2K];
#pragma unroll
  for (int n = 0; n < D2K; ++n) sim[n] = 0.0f;

  const _Float16* Pb = px + (long)b * CDIM * HWPIX;
  for (int c = 0; c < CDIM; ++c) {
    const _Float16* row = Pb + (long)c * HWPIX;
    __builtin_prefetch(row + HWPIX + p, 0, 0);   // next channel row
    float xc = (float)row[p];
#pragma unroll
    for (int n = 0; n < D2K; ++n)
      sim[n] += xc * (float)row[pn[n]];
  }

  const float scale = 0.08838834764831845f; // 1/sqrt(128)
  float mx = -1e30f;
#pragma unroll
  for (int n = 0; n < D2K; ++n) { sim[n] *= scale; mx = fmaxf(mx, sim[n]); }
  float ssum = 0.0f;
#pragma unroll
  for (int n = 0; n < D2K; ++n) { sim[n] = __expf(sim[n] - mx); ssum += sim[n]; }

  float sigma = sigma_p[0];
  float inv2s2 = 1.0f / (2.0f * sigma * sigma);
  float rs = 1.0f / ssum;

  float* Cb = comb + (long)b * D2K * HWPIX + p;
  _Float16* Kb = cat + (long)b * KCAT * HWPIX + p;
#pragma unroll
  for (int n = 0; n < D2K; ++n) {
    int ki = n / 7, kj = n % 7;
    float di = (float)(ki - 3) * (1.0f / 3.0f);
    float dj = (float)(kj - 3) * (1.0f / 3.0f);
    float sk = __expf(-(di * di + dj * dj) * inv2s2);
    float v = sim[n] * rs * sk;
    Cb[(long)n * HWPIX] = v;
    Kb[(long)n * HWPIX] = (_Float16)v;
  }
}

// ---------------------------------------------------------------------------
// combined = combined * (1 + sigmoid(fx)); normalize over 49
// ---------------------------------------------------------------------------
__global__ void gate_norm(const float* __restrict__ fx2,   // [B,64,HW], rows 0..48 valid
                          float* __restrict__ comb) {      // [B,49,HW] in/out
  int idx = blockIdx.x * blockDim.x + threadIdx.x;
  if (idx >= BQ * HWPIX) return;
  int b = idx / HWPIX, p = idx % HWPIX;
  const float* Fb = fx2 + (long)b * MFX * HWPIX + p;
  float* Cb = comb + (long)b * D2K * HWPIX + p;
  float cc[D2K];
  float ssum = 0.0f;
#pragma unroll
  for (int n = 0; n < D2K; ++n) {
    float f = Fb[(long)n * HWPIX];
    float gate = 1.0f / (1.0f + __expf(-f));
    float v = Cb[(long)n * HWPIX] * (1.0f + gate);
    cc[n] = v; ssum += v;
  }
  float rs = 1.0f / (ssum + 1e-7f);
#pragma unroll
  for (int n = 0; n < D2K; ++n) Cb[(long)n * HWPIX] = cc[n] * rs;
}

// ---------------------------------------------------------------------------
// out[c,p] = sum_n weight[n,p] * spatial[c, neighbor(p,n)]  -> f16 for op GEMM
// ---------------------------------------------------------------------------
__global__ void aggregate(const float* __restrict__ spat,  // [B,128,HW] f32
                          const float* __restrict__ comb,  // [B,49,HW]
                          _Float16* __restrict__ agg) {    // [B,128,HW] f16
  int idx = blockIdx.x * blockDim.x + threadIdx.x;
  if (idx >= BQ * HWPIX) return;
  int b = idx / HWPIX, p = idx % HWPIX;
  int h = p / WWD, w = p % WWD;

  int pn[D2K];
  float wv[D2K];
  const float* Cb = comb + (long)b * D2K * HWPIX + p;
#pragma unroll
  for (int n = 0; n < D2K; ++n) {
    int ki = n / 7, kj = n % 7;
    int hh = reflect_idx(h + ki - 3, HH);
    int ww = reflect_idx(w + kj - 3, WWD);
    pn[n] = hh * WWD + ww;
    wv[n] = Cb[(long)n * HWPIX];
  }
  const float* Sb = spat + (long)b * CDIM * HWPIX;
  _Float16* Ab = agg + (long)b * CDIM * HWPIX + p;
  for (int c = 0; c < CDIM; ++c) {
    const float* row = Sb + (long)c * HWPIX;
    __builtin_prefetch(row + HWPIX + p, 0, 0);   // next channel row
    float acc = 0.0f;
#pragma unroll
    for (int n = 0; n < D2K; ++n) acc += wv[n] * row[pn[n]];
    Ab[(long)c * HWPIX] = (_Float16)acc;
  }
}

// ---------------------------------------------------------------------------
extern "C" void kernel_launch(void* const* d_in, const int* in_sizes, int n_in,
                              void* d_out, int out_size, void* d_ws, size_t ws_size,
                              hipStream_t stream) {
  const float* spatial  = (const float*)d_in[0];
  const float* semantic = (const float*)d_in[1];
  const float* rp_w1 = (const float*)d_in[2];
  const float* rp_b1 = (const float*)d_in[3];
  const float* rp_lnw = (const float*)d_in[4];
  const float* rp_lnb = (const float*)d_in[5];
  const float* rp_w2 = (const float*)d_in[6];
  const float* rp_b2 = (const float*)d_in[7];
  const float* fx_w1 = (const float*)d_in[8];
  const float* fx_b1 = (const float*)d_in[9];
  const float* fx_lnw = (const float*)d_in[10];
  const float* fx_lnb = (const float*)d_in[11];
  const float* fx_w2 = (const float*)d_in[12];
  const float* fx_b2 = (const float*)d_in[13];
  const float* sigma = (const float*)d_in[14];
  const float* op_w1 = (const float*)d_in[15];
  const float* op_b1 = (const float*)d_in[16];
  const float* op_lnw = (const float*)d_in[17];
  const float* op_lnb = (const float*)d_in[18];
  const float* op_w2 = (const float*)d_in[19];
  const float* op_b2 = (const float*)d_in[20];
  float* out = (float*)d_out;

  // workspace carve-out (256B aligned)
  char* wp = (char*)d_ws;
  auto carve = [&](size_t bytes) -> void* {
    void* p = (void*)wp;
    wp += (bytes + 255) & ~(size_t)255;
    return p;
  };
  const long NHW = (long)BQ * HWPIX;
  _Float16* wh_rp1 = (_Float16*)carve(128 * 128 * sizeof(_Float16));
  _Float16* wh_rp2 = (_Float16*)carve(128 * 128 * sizeof(_Float16));
  _Float16* wh_fx1 = (_Float16*)carve(MFX * KCAT * sizeof(_Float16));
  _Float16* wh_fx2 = (_Float16*)carve(MFX * MFX * sizeof(_Float16));
  _Float16* wh_op1 = (_Float16*)carve(128 * 128 * sizeof(_Float16));
  _Float16* wh_op2 = (_Float16*)carve(128 * 128 * sizeof(_Float16));
  float* bp_rp1 = (float*)carve(128 * sizeof(float));
  float* bp_rp2 = (float*)carve(128 * sizeof(float));
  float* bp_fx1 = (float*)carve(MFX * sizeof(float));
  float* bp_fx2 = (float*)carve(MFX * sizeof(float));
  float* bp_op1 = (float*)carve(128 * sizeof(float));
  float* bp_op2 = (float*)carve(128 * sizeof(float));
  _Float16* sem_h  = (_Float16*)carve(NHW * CDIM * sizeof(_Float16));
  float*    t1     = (float*)   carve(NHW * CDIM * sizeof(float));    // reused
  _Float16* act_h  = (_Float16*)carve(NHW * CDIM * sizeof(_Float16)); // reused
  float*    projx  = (float*)   carve(NHW * CDIM * sizeof(float));
  _Float16* projxh = (_Float16*)carve(NHW * CDIM * sizeof(_Float16));
  _Float16* cat_h  = (_Float16*)carve(NHW * KCAT * sizeof(_Float16));
  float*    comb   = (float*)   carve(NHW * D2K * sizeof(float));
  float*    fx1f   = (float*)   carve(NHW * MFX * sizeof(float));
  _Float16* fxa_h  = (_Float16*)carve(NHW * MFX * sizeof(_Float16));
  float*    fx2f   = (float*)   carve(NHW * MFX * sizeof(float));
  _Float16* agg_h  = (_Float16*)carve(NHW * CDIM * sizeof(_Float16));

  const int TPB = 256;
  auto blocks = [](long n, int t) { return (int)((n + t - 1) / t); };

  // --- weights/biases -> padded
  pad_weight_f16<<<blocks(128 * 128, TPB), TPB, 0, stream>>>(rp_w1, 128, 128, wh_rp1, 128, 128);
  pad_weight_f16<<<blocks(128 * 128, TPB), TPB, 0, stream>>>(rp_w2, 128, 128, wh_rp2, 128, 128);
  pad_weight_f16<<<blocks(MFX * KCAT, TPB), TPB, 0, stream>>>(fx_w1, D2K, D2K + CDIM, wh_fx1, MFX, KCAT);
  pad_weight_f16<<<blocks(MFX * MFX, TPB), TPB, 0, stream>>>(fx_w2, D2K, D2K, wh_fx2, MFX, MFX);
  pad_weight_f16<<<blocks(128 * 128, TPB), TPB, 0, stream>>>(op_w1, 128, 128, wh_op1, 128, 128);
  pad_weight_f16<<<blocks(128 * 128, TPB), TPB, 0, stream>>>(op_w2, 128, 128, wh_op2, 128, 128);
  pad_bias_f32<<<1, 128, 0, stream>>>(rp_b1, 128, bp_rp1, 128);
  pad_bias_f32<<<1, 128, 0, stream>>>(rp_b2, 128, bp_rp2, 128);
  pad_bias_f32<<<1, MFX, 0, stream>>>(fx_b1, D2K, bp_fx1, MFX);
  pad_bias_f32<<<1, MFX, 0, stream>>>(fx_b2, D2K, bp_fx2, MFX);
  pad_bias_f32<<<1, 128, 0, stream>>>(op_b1, 128, bp_op1, 128);
  pad_bias_f32<<<1, 128, 0, stream>>>(op_b2, 128, bp_op2, 128);

  // --- semantic -> f16
  cvt_f32_f16<<<blocks(NHW * CDIM, TPB), TPB, 0, stream>>>(semantic, sem_h, (int)(NHW * CDIM));

  dim3 gblk(32, 4);                   // 4 waves
  dim3 ggrid(HWPIX / 16, 1, BQ);      // 16-column panels x batch
  int pxBlocks = blocks(NHW, TPB);

  // --- range_proj
  gemm_wmma<128, 2><<<ggrid, gblk, 0, stream>>>(wh_rp1, sem_h, bp_rp1, t1,
                                                (long)128 * HWPIX, (long)128 * HWPIX);
  ln2d_kernel<<<pxBlocks, TPB, 0, stream>>>(t1, rp_lnw, rp_lnb, act_h, 128, 128, 128, 1);
  gemm_wmma<128, 2><<<ggrid, gblk, 0, stream>>>(wh_rp2, act_h, bp_rp2, projx,
                                                (long)128 * HWPIX, (long)128 * HWPIX);
  cvt_f32_f16<<<blocks(NHW * CDIM, TPB), TPB, 0, stream>>>(projx, projxh, (int)(NHW * CDIM));

  // --- concat + sim/softmax
  build_concat<<<pxBlocks, TPB, 0, stream>>>(semantic, cat_h);
  sim_softmax<<<pxBlocks, TPB, 0, stream>>>(projxh, sigma, comb, cat_h);

  // --- fixup gating
  gemm_wmma<KCAT, 1><<<ggrid, gblk, 0, stream>>>(wh_fx1, cat_h, bp_fx1, fx1f,
                                                 (long)KCAT * HWPIX, (long)MFX * HWPIX);
  ln2d_kernel<<<pxBlocks, TPB, 0, stream>>>(fx1f, fx_lnw, fx_lnb, fxa_h, D2K, MFX, MFX, 1);
  gemm_wmma<MFX, 1><<<ggrid, gblk, 0, stream>>>(wh_fx2, fxa_h, bp_fx2, fx2f,
                                                (long)MFX * HWPIX, (long)MFX * HWPIX);
  gate_norm<<<pxBlocks, TPB, 0, stream>>>(fx2f, comb);

  // --- weighted aggregation
  aggregate<<<pxBlocks, TPB, 0, stream>>>(spatial, comb, agg_h);

  // --- output_proj (final GEMM writes d_out directly)
  gemm_wmma<128, 2><<<ggrid, gblk, 0, stream>>>(wh_op1, agg_h, bp_op1, t1,
                                                (long)128 * HWPIX, (long)128 * HWPIX);
  ln2d_kernel<<<pxBlocks, TPB, 0, stream>>>(t1, op_lnw, op_lnb, act_h, 128, 128, 128, 0);
  gemm_wmma<128, 2><<<ggrid, gblk, 0, stream>>>(wh_op2, act_h, bp_op2, out,
                                                (long)128 * HWPIX, (long)128 * HWPIX);
}
